// spectralNNDeepshared3_38362647888533
// MI455X (gfx1250) — compile-verified, matrix operand order checked by reading, not run on hardware
//
#include <hip/hip_runtime.h>
#include <hip/hip_bf16.h>

// ---------------------------------------------------------------------------
// Problem constants (from reference)
// ---------------------------------------------------------------------------
#define WIN    33      // 2*L+1
#define WIDTH  128
#define D_IN   64
#define MPTS   64
#define NDIM   2048
#define MDIM   2048
#define XICOLS 2080    // N + 2L
#define NCHUNK 16      // K-split factor for the conv GEMM (4 q-steps each)

typedef _Float16 v16h __attribute__((ext_vector_type(16)));
typedef _Float16 v8h  __attribute__((ext_vector_type(8)));
typedef _Float16 v2h  __attribute__((ext_vector_type(2)));
typedef float    v8f  __attribute__((ext_vector_type(8)));

// ---------------------------------------------------------------------------
// WMMA fragment helpers (16x16x32 f16, wave32 layouts per CDNA5 ISA 7.12.2)
//   A fragment: row = lane&15 ; K halves {klo..klo+7} and {16+klo..16+klo+7},
//   klo = 8*(lane>=16).  B fragment identical pattern on a [n][k] layout
//   (i.e. B supplied column-major / K-contiguous).
// ---------------------------------------------------------------------------
__device__ __forceinline__ v16h load_frag_f16(const _Float16* __restrict__ base,
                                              int ld, int lrow, int klo) {
  const _Float16* p = base + lrow * ld + klo;
  v8h lo = *(const v8h*)(p);        // K = klo .. klo+7        (16B load)
  v8h hi = *(const v8h*)(p + 16);   // K = 16+klo .. 16+klo+7  (16B load)
  return __builtin_shufflevector(lo, hi, 0, 1, 2, 3, 4, 5, 6, 7,
                                 8, 9, 10, 11, 12, 13, 14, 15);
}

__device__ __forceinline__ v16h load_frag_f32(const float* __restrict__ base,
                                              int ld, int lrow, int klo) {
  const float* p = base + lrow * ld + klo;
  v16h r;
#pragma unroll
  for (int t = 0; t < 8; ++t) r[t] = (_Float16)p[t];
#pragma unroll
  for (int t = 0; t < 8; ++t) r[8 + t] = (_Float16)p[16 + t];
  return r;
}

// fast sigmoid: v_exp_f32 + v_rcp_f32 (no IEEE division ladder)
__device__ __forceinline__ float sigf(float x) {
  return __builtin_amdgcn_rcpf(1.0f + __expf(-x));
}

// ---------------------------------------------------------------------------
// Kernel 1: per-window MLP stack.
//   h0[k,m] = sig( sum_l W0[j,k,l] * u[m,l] + b0[j,k] )          (128x64)
//   h[i+1][k,m] = sig( sum_l W[i,j][k,l] * h[i][l,m] + b[i,j,k]) (128x64)
//   u1_ws[j][m][k] = h_final  (f16, K=k contiguous for kernel 3's B operand)
// ---------------------------------------------------------------------------
__global__ __launch_bounds__(256) void mlp_kernel(
    const float* __restrict__ u,    // [64,64]
    const float* __restrict__ w0,   // [33,128,64]
    const float* __restrict__ b0,   // [33,128]
    const float* __restrict__ w,    // [2,33,128,128]
    const float* __restrict__ b,    // [2,33,128]
    _Float16* __restrict__ u1_ws)   // [33,64,128]
{
  __shared__ __align__(16) _Float16 u_lds[MPTS * D_IN];   // [m][l]  8 KB
  __shared__ __align__(16) _Float16 h_a[MPTS * WIDTH];    // [m][k] 16 KB
  __shared__ __align__(16) _Float16 h_b[MPTS * WIDTH];    // [m][k] 16 KB

  const int j    = blockIdx.x;
  const int tid  = threadIdx.x;
  const int wave = tid >> 5;
  const int lane = tid & 31;
  const int lrow = lane & 15;
  const int hs   = lane >> 4;
  const int klo  = hs * 8;

  for (int idx = tid; idx < MPTS * D_IN; idx += 256)
    u_lds[idx] = (_Float16)u[idx];
  __syncthreads();

  const int k0 = wave * 16;  // output-row tile for this wave

  // ----- stage 1: W0 (128x64) @ u^T -> h_a[m][k] -----
  {
    const float* w0j = w0 + j * WIDTH * D_IN;
    const float* b0j = b0 + j * WIDTH;
    for (int mt = 0; mt < 4; ++mt) {
      v8f c = {};
#pragma unroll
      for (int ks = 0; ks < 2; ++ks) {
        v16h a  = load_frag_f32(w0j + k0 * D_IN + ks * 32, D_IN, lrow, klo);
        v16h bf = load_frag_f16(u_lds + (mt * 16) * D_IN + ks * 32, D_IN, lrow, klo);
        c = __builtin_amdgcn_wmma_f32_16x16x32_f16(false, a, false, bf,
                                                   (short)0, c, false, false);
      }
      const int mm = mt * 16 + lrow;
#pragma unroll
      for (int r = 0; r < 8; r += 2) {
        const int kk = k0 + r + hs * 8;
        v2h pr = { (_Float16)sigf(c[r] + b0j[kk]),
                   (_Float16)sigf(c[r + 1] + b0j[kk + 1]) };
        *(v2h*)(h_a + mm * WIDTH + kk) = pr;
      }
    }
  }
  __syncthreads();

  // ----- hidden layers: W[i,j] (128x128) @ h -----
  _Float16* hcur = h_a;
  _Float16* hnxt = h_b;
  for (int layer = 0; layer < 2; ++layer) {
    const float* wl = w + ((size_t)(layer * WIN + j)) * WIDTH * WIDTH;
    const float* bl = b + (layer * WIN + j) * WIDTH;
    for (int mt = 0; mt < 4; ++mt) {
      v8f c = {};
#pragma unroll
      for (int ks = 0; ks < 4; ++ks) {
        v16h a  = load_frag_f32(wl + k0 * WIDTH + ks * 32, WIDTH, lrow, klo);
        v16h bf = load_frag_f16(hcur + (mt * 16) * WIDTH + ks * 32, WIDTH, lrow, klo);
        c = __builtin_amdgcn_wmma_f32_16x16x32_f16(false, a, false, bf,
                                                   (short)0, c, false, false);
      }
      const int mm = mt * 16 + lrow;
#pragma unroll
      for (int r = 0; r < 8; r += 2) {
        const int kk = k0 + r + hs * 8;
        v2h pr = { (_Float16)sigf(c[r] + bl[kk]),
                   (_Float16)sigf(c[r + 1] + bl[kk + 1]) };
        *(v2h*)(hnxt + mm * WIDTH + kk) = pr;
      }
    }
    __syncthreads();
    _Float16* t = hcur; hcur = hnxt; hnxt = t;
  }

  for (int idx = tid; idx < MPTS * WIDTH; idx += 256)
    u1_ws[(size_t)j * MPTS * WIDTH + idx] = hcur[idx];
}

// ---------------------------------------------------------------------------
// Kernel 2: xi (f32 [2048,2080]) -> xiT (f16 [2080,2048]) tiled transpose.
// ---------------------------------------------------------------------------
__global__ __launch_bounds__(256) void xit_kernel(
    const float* __restrict__ xi,   // [2048, 2080]
    _Float16* __restrict__ xiT)     // [2080, 2048]
{
  __shared__ _Float16 tile[32][33];
  const int tx = threadIdx.x & 31;
  const int ty = threadIdx.x >> 5;
  const int c0 = blockIdx.x * 32;
  const int q0 = blockIdx.y * 32;

#pragma unroll
  for (int it = 0; it < 4; ++it) {
    const int q = q0 + ty + it * 8;
    tile[ty + it * 8][tx] = (_Float16)xi[(size_t)q * XICOLS + (c0 + tx)];
  }
  __syncthreads();
#pragma unroll
  for (int it = 0; it < 4; ++it) {
    const int ccol = c0 + ty + it * 8;
    xiT[(size_t)ccol * MDIM + (q0 + tx)] = tile[tx][ty + it * 8];
  }
}

// ---------------------------------------------------------------------------
// Kernel 3: G[q,j,p] = sig( sum_k Wf[q,j,k] * u1[j,k,p] + bf[q,j] )
// Output as f16 G_ws[j][p][q] (K=q contiguous for kernel 4's B operand).
// ---------------------------------------------------------------------------
__global__ __launch_bounds__(256) void gmat_kernel(
    const float* __restrict__ wf,        // [2048,33,128]
    const float* __restrict__ bfin,      // [2048,33]
    const _Float16* __restrict__ u1_ws,  // [33,64,128]
    _Float16* __restrict__ G_ws)         // [33,64,2048]  ([j][p][q])
{
  __shared__ __align__(16) _Float16 u1j[MPTS * WIDTH];  // 16 KB
  const int j    = blockIdx.y;
  const int qblk = blockIdx.x;
  const int tid  = threadIdx.x;
  const int wave = tid >> 5;
  const int lane = tid & 31;
  const int lrow = lane & 15;
  const int hs   = lane >> 4;
  const int klo  = hs * 8;

  for (int idx = tid; idx < MPTS * WIDTH; idx += 256)
    u1j[idx] = u1_ws[(size_t)j * MPTS * WIDTH + idx];
  __syncthreads();

  const int q0  = qblk * 128 + wave * 16;
  const int ldA = WIN * WIDTH;  // 4224 floats between consecutive q rows

  for (int pt = 0; pt < 4; ++pt) {
    v8f c = {};
#pragma unroll
    for (int ks = 0; ks < 4; ++ks) {
      v16h a  = load_frag_f32(wf + ((size_t)q0 * WIN + j) * WIDTH + ks * 32,
                              ldA, lrow, klo);
      v16h bf = load_frag_f16(u1j + (pt * 16) * WIDTH + ks * 32, WIDTH, lrow, klo);
      c = __builtin_amdgcn_wmma_f32_16x16x32_f16(false, a, false, bf,
                                                 (short)0, c, false, false);
    }
    const int p = pt * 16 + lrow;
#pragma unroll
    for (int r = 0; r < 8; r += 2) {
      const int q = q0 + r + hs * 8;
      v2h pr = { (_Float16)sigf(c[r]     + bfin[q * WIN + j]),
                 (_Float16)sigf(c[r + 1] + bfin[(q + 1) * WIN + j]) };
      *(v2h*)(G_ws + ((size_t)j * MPTS + p) * MDIM + q) = pr;
    }
  }
}

// ---------------------------------------------------------------------------
// Kernel 4: out[n,p] = sum_j sum_q xiT[n+j, q] * G[q, j, p]
// GEMM N=2048, P=64, K=33*2048.
//   block = 8 waves over 256 n-rows; wave = 2 n-tiles x 4 p-tiles
//   (8 accumulators -> B fragments reused 2x in regs, A reused 4x,
//    1.5 ds_load_b128 per WMMA).
//   per q-step: A slab (288 rows x 32 q, covers all 33 shifts) staged once;
//   G tiles double-buffered: one barrier per j, staging overlaps compute.
// grid = (8 n-blocks, NCHUNK K-chunks); deterministic partial reduction.
// ---------------------------------------------------------------------------
#define APAD 40   // padded LDS row stride (halves): spreads 16B frags over banks

__global__ __launch_bounds__(256) void conv_kernel(
    const _Float16* __restrict__ xiT,   // [2080, 2048]
    const _Float16* __restrict__ G_ws,  // [33,64,2048]
    float* __restrict__ part)           // [NCHUNK][2048*64]
{
  __shared__ __align__(16) _Float16 a_tile[288 * APAD];     // 23.0 KB
  __shared__ __align__(16) _Float16 g_tile[2][64 * APAD];   // 2 x 5.1 KB

  const int tid  = threadIdx.x;
  const int wave = tid >> 5;
  const int lane = tid & 31;
  const int lrow = lane & 15;
  const int hs   = lane >> 4;
  const int klo  = hs * 8;

  const int n0    = blockIdx.x * 256;
  const int chunk = blockIdx.y;  // 4 q-steps (128 q values) per chunk

  v8f acc[2][4] = {};

  for (int qs = chunk * 4; qs < chunk * 4 + 4; ++qs) {
    const int q0 = qs * 32;

    // stage A slab: rows n0..n0+287, cols q0..q0+31 (1152 16B chunks)
    for (int idx = tid; idx < 1152; idx += 256) {
      const int row = idx >> 2, prt = idx & 3;
      *(v8h*)(a_tile + row * APAD + prt * 8) =
          *(const v8h*)(xiT + (size_t)(n0 + row) * MDIM + q0 + prt * 8);
    }
    // stage G(j=0) into buffer 0: one 16B chunk per thread
    {
      const int p = tid >> 2, prt = tid & 3;
      *(v8h*)(g_tile[0] + p * APAD + prt * 8) =
          *(const v8h*)(G_ws + ((size_t)0 * MPTS + p) * MDIM + q0 + prt * 8);
    }

    for (int j = 0; j < WIN; ++j) {
      __syncthreads();  // G(j) staged; prior reads of the other buffer done
      if (j + 1 < WIN) {  // prefetch G(j+1) while computing on G(j)
        const int p = tid >> 2, prt = tid & 3;
        *(v8h*)(g_tile[(j + 1) & 1] + p * APAD + prt * 8) =
            *(const v8h*)(G_ws + ((size_t)(j + 1) * MPTS + p) * MDIM + q0 + prt * 8);
      }
      const _Float16* gb = g_tile[j & 1];
      v16h a0 = load_frag_f16(a_tile + (wave * 16 + j) * APAD,       APAD, lrow, klo);
      v16h a1 = load_frag_f16(a_tile + (128 + wave * 16 + j) * APAD, APAD, lrow, klo);
#pragma unroll
      for (int pt = 0; pt < 4; ++pt) {
        v16h b = load_frag_f16(gb + pt * 16 * APAD, APAD, lrow, klo);
        acc[0][pt] = __builtin_amdgcn_wmma_f32_16x16x32_f16(
            false, a0, false, b, (short)0, acc[0][pt], false, false);
        acc[1][pt] = __builtin_amdgcn_wmma_f32_16x16x32_f16(
            false, a1, false, b, (short)0, acc[1][pt], false, false);
      }
    }
    __syncthreads();  // protect a_tile / g_tile[0] before next q-step staging
  }

  float* dst = part + (size_t)chunk * (NDIM * MPTS);
#pragma unroll
  for (int t = 0; t < 2; ++t) {
    const int nbase = n0 + t * 128 + wave * 16 + hs * 8;
#pragma unroll
    for (int pt = 0; pt < 4; ++pt) {
      const int p = pt * 16 + lrow;
#pragma unroll
      for (int r = 0; r < 8; ++r)
        dst[(size_t)(nbase + r) * MPTS + p] = acc[t][pt][r];
    }
  }
}

// ---------------------------------------------------------------------------
// Kernel 5: deterministic NCHUNK-way reduction of K-split partials.
// ---------------------------------------------------------------------------
__global__ __launch_bounds__(256) void reduce_kernel(
    const float* __restrict__ part, float* __restrict__ out)
{
  const int i = blockIdx.x * 256 + threadIdx.x;
  const int stride = NDIM * MPTS;  // 131072
  if (i < stride) {
    float s = 0.0f;
#pragma unroll
    for (int c = 0; c < NCHUNK; ++c)  // fixed order -> deterministic
      s += part[i + (size_t)c * stride];
    out[i] = s;
  }
}

// ---------------------------------------------------------------------------
// Launcher
// ---------------------------------------------------------------------------
extern "C" void kernel_launch(void* const* d_in, const int* in_sizes, int n_in,
                              void* d_out, int out_size, void* d_ws, size_t ws_size,
                              hipStream_t stream) {
  (void)in_sizes; (void)n_in; (void)out_size; (void)ws_size;

  const float* u   = (const float*)d_in[0];
  const float* w0  = (const float*)d_in[1];
  const float* b0  = (const float*)d_in[2];
  const float* w   = (const float*)d_in[3];
  const float* b   = (const float*)d_in[4];
  const float* wf  = (const float*)d_in[5];
  const float* bfn = (const float*)d_in[6];
  const float* xi  = (const float*)d_in[7];
  float* out = (float*)d_out;

  // workspace layout (bytes, 16B-aligned)
  char* ws = (char*)d_ws;
  _Float16* u1_ws = (_Float16*)(ws);                       //   540,672 B
  _Float16* xiT   = (_Float16*)(ws + 540672);              // 8,519,680 B
  _Float16* G_ws  = (_Float16*)(ws + 540672 + 8519680);    // 8,650,752 B
  float*    part  = (float*)(ws + 540672 + 8519680 + 8650752); // 8,388,608 B

  mlp_kernel<<<dim3(WIN), dim3(256), 0, stream>>>(u, w0, b0, w, b, u1_ws);
  xit_kernel<<<dim3(XICOLS / 32, MDIM / 32), dim3(256), 0, stream>>>(xi, xiT);
  gmat_kernel<<<dim3(16, WIN), dim3(256), 0, stream>>>(wf, bfn, u1_ws, G_ws);
  conv_kernel<<<dim3(NDIM / 256, NCHUNK), dim3(256), 0, stream>>>(xiT, G_ws, part);
  reduce_kernel<<<dim3((NDIM * MPTS) / 256), dim3(256), 0, stream>>>(part, out);
}
